// minimalGRU_24111946400152
// MI455X (gfx1250) — matrix operand order repaired and checked by reading, not compiled, not run on hardware
//
#include <hip/hip_runtime.h>
#include <hip/hip_bf16.h>
#include <math.h>
#include <stdint.h>

// ---------------------------------------------------------------------------
// Minimal 2-layer GRU for MI455X (gfx1250), wave32 + WMMA f16 matrix cores.
//   T=2000, B=64, D=H=512
// Plan:
//   1) convert x and all 8 weight matrices f32 -> f16
//   2) big WMMA GEMMs: WH = X@Wh^T + bh, WZ = X@Wz^T + bz   (per layer)
//   3) sequential recurrence: ONE persistent 32-wave workgroup on one WGP.
//      - h and z*h operands resident in LDS (gfx1250: 320KB/WGP)
//      - wave w owns output n-tile w => each U fragment loaded exactly once
//      - wz[t], wh[t] staged into LDS with global_load_async_to_lds_b128,
//        overlapped with the phase-A WMMA k-loop (s_wait_asynccnt fence)
// ---------------------------------------------------------------------------

#define T_STEPS 2000
#define BB      64
#define DD      512
#define HH      512
#define LDH     520   // padded LDS row stride in halves (16B-aligned, bank-skewed)

typedef __attribute__((ext_vector_type(16))) _Float16 v16h;
typedef __attribute__((ext_vector_type(8)))  _Float16 v8h;
typedef __attribute__((ext_vector_type(8)))  float    v8f;

union AFrag { v16h v; v8h h[2]; };

// A-matrix fragment (16x32 f16; this code's contiguous k order, same as B): lane L holds row L%16;
// elems 0..15 = k0 + (L/16)*16 + {0..15}.
__device__ __forceinline__ v16h load_a_frag(const _Float16* __restrict__ tile,
                                            int ldk, int lane) {
  const int row  = lane & 15;
  const int koff = (lane >> 4) << 4;
  const _Float16* p = tile + row * ldk + koff;
  AFrag f;
  f.h[0] = *(const v8h*)(p);
  f.h[1] = *(const v8h*)(p + 8);
  return f.v;
}

// B-matrix fragment (32x16 f16): lane L holds output-col L%16 (= row of W/U
// in row-major [N,K]); 16 contiguous halves at k0 + (L/16)*16.  One 32B load.
__device__ __forceinline__ v16h load_b_frag(const _Float16* __restrict__ tile,
                                            int ldk, int lane) {
  const int col  = lane & 15;
  const int koff = (lane >> 4) << 4;
  return *(const v16h*)(tile + col * ldk + koff);
}

__device__ __forceinline__ v8f wmma_f16(v16h a, v16h b, v8f c) {
  return __builtin_amdgcn_wmma_f32_16x16x32_f16(false, a, false, b,
                                                (short)0, c, false, false);
}

__device__ __forceinline__ float sigmoidf_fast(float x) {
  return 1.0f / (1.0f + __expf(-x));
}

// ---------------------------------------------------------------------------
__global__ void gru_f32_to_f16_kernel(const float* __restrict__ in,
                                      _Float16* __restrict__ out, size_t n) {
  size_t i = (size_t)blockIdx.x * blockDim.x + threadIdx.x;
  if (i < n) out[i] = (_Float16)in[i];
}

// Y[M,512] = A[M,512] @ W[512,512]^T + bias ; output f16.
// 256 threads = 8 waves; block computes 64x64; wave = 1 A-row-tile x 2 N-tiles.
__global__ __launch_bounds__(256)
void gru_proj_gemm_kernel(const _Float16* __restrict__ A,
                          const _Float16* __restrict__ W,   // [N,K] row-major
                          const float*    __restrict__ bias,
                          _Float16*       __restrict__ Y) {
  const int K = 512, N = 512;
  const int tid  = threadIdx.x;
  const int lane = tid & 31;
  const int wave = tid >> 5;          // 0..7
  const int mt   = wave & 3;          // 0..3
  const int np   = wave >> 2;         // 0..1
  const int m0 = blockIdx.x * 64 + mt * 16;
  const int n0 = blockIdx.y * 64 + (np * 2) * 16;
  const int n1 = n0 + 16;

  v8f acc0 = {}, acc1 = {};
  const _Float16* arow = A + (size_t)m0 * K;
  const _Float16* w0   = W + (size_t)n0 * K;
  const _Float16* w1   = W + (size_t)n1 * K;

  for (int k0 = 0; k0 < K; k0 += 32) {
    v16h a  = load_a_frag(arow + k0, K, lane);
    v16h b0 = load_b_frag(w0 + k0, K, lane);
    v16h b1 = load_b_frag(w1 + k0, K, lane);
    acc0 = wmma_f16(a, b0, acc0);
    acc1 = wmma_f16(a, b1, acc1);
  }

  const int crow = m0 + ((lane >> 4) << 3);
  const int ccol = lane & 15;
  const float bia0 = bias[n0 + ccol];
  const float bia1 = bias[n1 + ccol];
#pragma unroll
  for (int v = 0; v < 8; ++v) {
    const size_t r = (size_t)(crow + v) * N;
    Y[r + n0 + ccol] = (_Float16)(acc0[v] + bia0);
    Y[r + n1 + ccol] = (_Float16)(acc1[v] + bia1);
  }
}

// Persistent single-workgroup recurrence over all T steps of one layer.
// 1024 threads = 32 waves on one WGP.  LDS: h, z*h (padded) + wz/wh stage.
// Wave w owns N-tile w (cols 16w..16w+15) and all four M-tiles (B=64 rows):
// every U fragment is loaded exactly once per step (minimal L2 traffic).
__global__ __launch_bounds__(1024)
void gru_recurrence_kernel(const _Float16* __restrict__ WH,  // [T,B,H] f16
                           const _Float16* __restrict__ WZ,  // [T,B,H] f16
                           const _Float16* __restrict__ Uh,  // [H,H] f16
                           const _Float16* __restrict__ Uz,  // [H,H] f16
                           _Float16* __restrict__ out16,     // [T,B,H] f16
                           float*    __restrict__ out32,     // [T,B,H] or null
                           int write32) {
  extern __shared__ _Float16 smem[];
  _Float16* sh  = smem;                       // h state [BB][LDH]
  _Float16* szh = smem + BB * LDH;            // z*h     [BB][LDH]
  _Float16* swz = smem + 2 * BB * LDH;        // wz[t]   [BB][HH]
  _Float16* swh = swz + BB * HH;              // wh[t]   [BB][HH]

  const int tid  = threadIdx.x;
  const int lane = tid & 31;
  const int wave = tid >> 5;                  // n-tile id 0..31
  const int n0   = wave * 16;
  const int ccol = n0 + (lane & 15);
  const int rbase = (lane >> 4) << 3;

  // LDS byte offsets of the staging buffers (dynamic LDS starts at 0)
  const unsigned swz_off = (unsigned)((const char*)swz - (const char*)smem);
  const unsigned swh_off = (unsigned)((const char*)swh - (const char*)smem);

  // h0 = 0
  for (int i = tid; i < BB * LDH; i += 1024) sh[i] = (_Float16)0.0f;
  __syncthreads();

  for (int t = 0; t < T_STEPS; ++t) {
    const _Float16* wzt = WZ + (size_t)t * BB * HH;
    const _Float16* wht = WH + (size_t)t * BB * HH;

    // ---- async-stage wz[t], wh[t] into LDS (16B per lane per beat) --------
#pragma unroll
    for (int kb = 0; kb < 4; ++kb) {
      const unsigned boff = ((unsigned)tid + (unsigned)kb * 1024u) * 16u;
      unsigned long long gz = (unsigned long long)(uintptr_t)((const char*)wzt + boff);
      unsigned long long gh = (unsigned long long)(uintptr_t)((const char*)wht + boff);
      unsigned dz = swz_off + boff;
      unsigned dh = swh_off + boff;
      asm volatile("global_load_async_to_lds_b128 %0, %1, off"
                   :: "v"(dz), "v"(gz) : "memory");
      asm volatile("global_load_async_to_lds_b128 %0, %1, off"
                   :: "v"(dh), "v"(gh) : "memory");
    }

    // ---- phase A GEMM: G = h @ Uz^T  (4 M-tiles x this wave's N-tile) -----
    v8f acc[4] = {};
    for (int k0 = 0; k0 < HH; k0 += 32) {
      const _Float16* bp = Uz + (size_t)n0 * HH + k0;
      if (k0 + 32 < HH) __builtin_prefetch(bp + 32, 0, 1);  // global_prefetch_b8
      v16h b = load_b_frag(bp, HH, lane);
#pragma unroll
      for (int m = 0; m < 4; ++m) {
        v16h a = load_a_frag(sh + (m * 16) * LDH + k0, LDH, lane);
        acc[m] = wmma_f16(a, b, acc[m]);
      }
    }
    // DMA fence: this wave's async loads done, then make all waves' visible
    asm volatile("s_wait_asynccnt 0x0" ::: "memory");
    __syncthreads();

    // ---- epilogue A: z = sigmoid(wz + G); szh = z*h (owner-exclusive) ----
    float zreg[4][8];
#pragma unroll
    for (int m = 0; m < 4; ++m) {
#pragma unroll
      for (int v = 0; v < 8; ++v) {
        const int row = m * 16 + rbase + v;
        const float wzv = (float)swz[row * HH + ccol];
        const float z   = sigmoidf_fast(wzv + acc[m][v]);
        zreg[m][v] = z;
        const float hold = (float)sh[row * LDH + ccol];
        szh[row * LDH + ccol] = (_Float16)(z * hold);
      }
    }
    __syncthreads();

    // ---- phase B: hc = tanh(wh + (z*h) @ Uh^T); h = z*h + (1-z)*hc -------
#pragma unroll
    for (int m = 0; m < 4; ++m) acc[m] = (v8f){};
    for (int k0 = 0; k0 < HH; k0 += 32) {
      const _Float16* bp = Uh + (size_t)n0 * HH + k0;
      if (k0 + 32 < HH) __builtin_prefetch(bp + 32, 0, 1);
      v16h b = load_b_frag(bp, HH, lane);
#pragma unroll
      for (int m = 0; m < 4; ++m) {
        v16h a = load_a_frag(szh + (m * 16) * LDH + k0, LDH, lane);
        acc[m] = wmma_f16(a, b, acc[m]);
      }
    }
#pragma unroll
    for (int m = 0; m < 4; ++m) {
#pragma unroll
      for (int v = 0; v < 8; ++v) {
        const int row = m * 16 + rbase + v;
        const float whv  = (float)swh[row * HH + ccol];
        const float hc   = tanhf(whv + acc[m][v]);
        const float z    = zreg[m][v];
        const float hold = (float)sh[row * LDH + ccol];
        const float hn   = z * hold + (1.0f - z) * hc;
        sh[row * LDH + ccol] = (_Float16)hn;                      // owner-exclusive
        out16[(size_t)t * BB * HH + row * HH + ccol] = (_Float16)hn;
        if (write32) out32[(size_t)t * BB * HH + row * HH + ccol] = hn;
      }
    }
    __syncthreads();  // h updated + swz/swh reads done before next step's DMA
  }
}

// ---------------------------------------------------------------------------
extern "C" void kernel_launch(void* const* d_in, const int* in_sizes, int n_in,
                              void* d_out, int out_size, void* d_ws, size_t ws_size,
                              hipStream_t stream) {
  (void)in_sizes; (void)n_in; (void)out_size; (void)ws_size;

  const float* x   = (const float*)d_in[0];
  const float* Wh0 = (const float*)d_in[1];
  const float* bh0 = (const float*)d_in[2];
  const float* Wz0 = (const float*)d_in[3];
  const float* bz0 = (const float*)d_in[4];
  const float* Uh0 = (const float*)d_in[5];
  const float* Uz0 = (const float*)d_in[6];
  const float* Wh1 = (const float*)d_in[7];
  const float* bh1 = (const float*)d_in[8];
  const float* Wz1 = (const float*)d_in[9];
  const float* bz1 = (const float*)d_in[10];
  const float* Uh1 = (const float*)d_in[11];
  const float* Uz1 = (const float*)d_in[12];
  float* out = (float*)d_out;

  const size_t nTBH = (size_t)T_STEPS * BB * HH;   // 65,536,000
  const size_t nW   = (size_t)HH * DD;             // 262,144

  // Workspace carve-up (256B aligned)
  char*  base = (char*)d_ws;
  size_t off  = 0;
  auto take = [&](size_t bytes) -> char* {
    char* p = base + off;
    off += (bytes + 255) & ~(size_t)255;
    return p;
  };
  _Float16* xh    = (_Float16*)take(nTBH * 2);   // x (layer0 input) in f16
  _Float16* WHbuf = (_Float16*)take(nTBH * 2);   // wh projections
  _Float16* WZbuf = (_Float16*)take(nTBH * 2);   // wz projections
  _Float16* hseq  = (_Float16*)take(nTBH * 2);   // layer output f16 (next input)
  _Float16* Wh0h  = (_Float16*)take(nW * 2);
  _Float16* Wz0h  = (_Float16*)take(nW * 2);
  _Float16* Uh0h  = (_Float16*)take(nW * 2);
  _Float16* Uz0h  = (_Float16*)take(nW * 2);
  _Float16* Wh1h  = (_Float16*)take(nW * 2);
  _Float16* Wz1h  = (_Float16*)take(nW * 2);
  _Float16* Uh1h  = (_Float16*)take(nW * 2);
  _Float16* Uz1h  = (_Float16*)take(nW * 2);

  // 1) precision conversions
  {
    const int thr = 256;
    gru_f32_to_f16_kernel<<<(unsigned)((nTBH + thr - 1) / thr), thr, 0, stream>>>(x, xh, nTBH);
    const unsigned wg = (unsigned)((nW + thr - 1) / thr);
    gru_f32_to_f16_kernel<<<wg, thr, 0, stream>>>(Wh0, Wh0h, nW);
    gru_f32_to_f16_kernel<<<wg, thr, 0, stream>>>(Wz0, Wz0h, nW);
    gru_f32_to_f16_kernel<<<wg, thr, 0, stream>>>(Uh0, Uh0h, nW);
    gru_f32_to_f16_kernel<<<wg, thr, 0, stream>>>(Uz0, Uz0h, nW);
    gru_f32_to_f16_kernel<<<wg, thr, 0, stream>>>(Wh1, Wh1h, nW);
    gru_f32_to_f16_kernel<<<wg, thr, 0, stream>>>(Wz1, Wz1h, nW);
    gru_f32_to_f16_kernel<<<wg, thr, 0, stream>>>(Uh1, Uh1h, nW);
    gru_f32_to_f16_kernel<<<wg, thr, 0, stream>>>(Uz1, Uz1h, nW);
  }

  const dim3 gemmGrid((T_STEPS * BB) / 64, HH / 64);     // (2000, 8)
  // LDS: h + z*h (padded) + wz/wh staging = 264,192 bytes (<= 320KB WGP LDS)
  const size_t ldsBytes =
      ((size_t)2 * BB * LDH + (size_t)2 * BB * HH) * sizeof(_Float16);

  // 2) layer 0: projections + recurrence (writes hseq f16)
  gru_proj_gemm_kernel<<<gemmGrid, 256, 0, stream>>>(xh, Wh0h, bh0, WHbuf);
  gru_proj_gemm_kernel<<<gemmGrid, 256, 0, stream>>>(xh, Wz0h, bz0, WZbuf);
  gru_recurrence_kernel<<<1, 1024, ldsBytes, stream>>>(WHbuf, WZbuf, Uh0h, Uz0h,
                                                       hseq, nullptr, 0);

  // 3) layer 1: projections + recurrence (writes d_out f32)
  gru_proj_gemm_kernel<<<gemmGrid, 256, 0, stream>>>(hseq, Wh1h, bh1, WHbuf);
  gru_proj_gemm_kernel<<<gemmGrid, 256, 0, stream>>>(hseq, Wz1h, bz1, WZbuf);
  gru_recurrence_kernel<<<1, 1024, ldsBytes, stream>>>(WHbuf, WZbuf, Uh1h, Uz1h,
                                                       hseq, out, 1);
}
